// AudioDeviceModel_46514495816284
// MI455X (gfx1250) — matrix operand order repaired and checked by reading, not compiled, not run on hardware
//
#include <hip/hip_runtime.h>

// WaveNet-style stack, MI455X (gfx1250, wave32).
// One workgroup (512 threads = 16 waves) per batch element. Entire needed
// activation window (last 4352 timesteps >= 4092 receptive field + 128 outputs)
// lives in LDS as f16, ping-pong across layers. Each 16-timestep tile uses two
// v_wmma_f32_16x16x32_f16: (1) dilated conv as [16 x 24pad32]x[32 x 16] GEMM,
// (2) residual 1x1 as [16 x 8pad32]x[32 x 16] GEMM with C = x + io_b.
// hstage transpose is wave-private: LDS ops are in-order per wave (ISA: DScnt
// in-order), so only a compiler scheduling fence is needed, not __syncthreads.

#define T_FULL   8192
#define N_WIN    4352            // >= 128 + 4092, multiple of 128
#define N_TILES  (N_WIN / 16)    // 272 -> 17 tiles per wave (16 waves)
#define NWAVES   16
#define NTHREADS (NWAVES * 32)
#define NLAYERS  20

typedef __attribute__((ext_vector_type(16))) _Float16 v16h;
typedef __attribute__((ext_vector_type(8)))  float    v8f;

union Frag16 {
    v16h      v;
    _Float16  h[16];
    float4    f4[2];
};

__global__ __launch_bounds__(NTHREADS)
void wavenet_kernel(const float* __restrict__ audio,
                    const float* __restrict__ ctrl,
                    const float* __restrict__ c_w0,   // [3,1,8]
                    const float* __restrict__ c_ws,   // [19,3,8,8]
                    const float* __restrict__ c_b,    // [20,8]
                    const float* __restrict__ ctrl_w, // [20,1,1]
                    const float* __restrict__ ctrl_b, // [20,1]
                    const float* __restrict__ io_w,   // [19,8,8]
                    const float* __restrict__ io_b,   // [19,8]
                    const float* __restrict__ mix_w,  // [160,1]
                    const float* __restrict__ mix_b,  // [1]
                    float* __restrict__ out)          // [B,128]
{
    __shared__ __align__(16) _Float16 xA[N_WIN * 8];          // 69632 B
    __shared__ __align__(16) _Float16 xB[N_WIN * 8];          // 69632 B
    __shared__ float ctrls[N_WIN];                            // 17408 B
    __shared__ __align__(16) _Float16 hstage[NWAVES][16 * 8]; //  4096 B
    __shared__ float oacc[128];
    __shared__ __align__(16) _Float16 zrow[8];

    const int tid  = threadIdx.x;
    const int lane = tid & 31;
    const int wave = tid >> 5;
    const int b    = blockIdx.x;
    const int base = b * T_FULL + (T_FULL - N_WIN);

    // ---- stage inputs: replicate 1-ch audio into all 8 channels ----
    for (int t = tid; t < N_WIN; t += NTHREADS) {
        _Float16 ah = (_Float16)audio[base + t];
        #pragma unroll
        for (int c = 0; c < 8; ++c) xA[t * 8 + c] = ah;
        ctrls[t] = ctrl[base + t];
    }
    if (tid < 128) oacc[tid] = 0.0f;
    if (tid < 8)   zrow[tid] = (_Float16)0.0f;
    __syncthreads();

    _Float16* cur = xA;
    _Float16* nxt = xB;

    const int  co     = lane & 15;   // column (output channel) of this lane
    const bool lanelo = lane < 16;
    const bool realc  = co < 8;

    for (int i = 0; i < NLAYERS; ++i) {
        const int   d    = 1 << (i % 10);         // DILATIONS pattern
        const bool  last = (i == NLAYERS - 1);
        const float cw   = ctrl_w[i];
        const float cbb  = ctrl_b[i];
        const float bias = realc ? c_b[i * 8 + co]  : 0.0f;
        const float mw   = realc ? mix_w[i * 8 + co] : 0.0f;
        const float iob  = (!last && realc) ? io_b[i * 8 + co] : 0.0f;

        // ---- conv weight fragment (B: 32x16, K = tap*8+cin), regs all layer ----
        // B layout: lanes 0-15 elems0-7 K=0..7, elems8-15 K=16..23; lanes 16-31 elems0-7 K=8..15, elems8-15 K=24..31
        Frag16 bc;
        #pragma unroll
        for (int e = 0; e < 16; ++e) bc.h[e] = (_Float16)0.0f;
        if (realc) {
            if (i == 0) {                     // 1 input channel (cin==0 only)
                if (lanelo) {
                    bc.h[0] = (_Float16)c_w0[0 * 8 + co];   // K0  = tap0,cin0
                    bc.h[8] = (_Float16)c_w0[2 * 8 + co];   // K16 = tap2,cin0
                } else {
                    bc.h[0] = (_Float16)c_w0[1 * 8 + co];   // K8  = tap1,cin0
                }
            } else {
                const float* w = c_ws + (size_t)(i - 1) * 192;
                if (lanelo) {
                    #pragma unroll
                    for (int e = 0; e < 8; ++e) {
                        bc.h[e]     = (_Float16)w[(0 * 8 + e) * 8 + co]; // tap0
                        bc.h[8 + e] = (_Float16)w[(2 * 8 + e) * 8 + co]; // tap2
                    }
                } else {
                    #pragma unroll
                    for (int e = 0; e < 8; ++e) {
                        bc.h[e]     = (_Float16)w[(1 * 8 + e) * 8 + co]; // tap1
                    }
                }
            }
        }
        // ---- residual 1x1 weight fragment (B: 32x16, K = input channel o) ----
        Frag16 bio;
        #pragma unroll
        for (int e = 0; e < 16; ++e) bio.h[e] = (_Float16)0.0f;
        if (!last && realc && lanelo) {
            const float* w = io_w + (size_t)i * 64;
            #pragma unroll
            for (int e = 0; e < 8; ++e) bio.h[e] = (_Float16)w[e * 8 + co];
        }

        for (int tile = wave; tile < N_TILES; tile += NWAVES) { // 17 iters/wave
            const int t0 = tile * 16;
            const int m  = co;          // A-row (timestep within tile)
            const int t  = t0 + m;

            // ---- A fragment for conv: row m = [x(t-2d)|x(t-d)|x(t)|0] ----
            // lane m<16: elems0-7 = K0..7 (tap0), elems8-15 = K16..23 (tap2)
            // lane m+16: elems0-7 = K8..15 (tap1), elems8-15 = 0
            Frag16 a;
            if (lanelo) {
                const int i0 = t - 2 * d;
                const _Float16* p0 = (i0 >= 0) ? (cur + i0 * 8) : zrow;
                a.f4[0] = *reinterpret_cast<const float4*>(p0);
                a.f4[1] = *reinterpret_cast<const float4*>(cur + t * 8);
            } else {
                const int i1 = t - d;
                const _Float16* p1 = (i1 >= 0) ? (cur + i1 * 8) : zrow;
                a.f4[0] = *reinterpret_cast<const float4*>(p1);
                a.f4[1] = make_float4(0.f, 0.f, 0.f, 0.f);
            }

            v8f dacc = {};
            dacc = __builtin_amdgcn_wmma_f32_16x16x32_f16(
                       false, a.v, false, bc.v, (short)0, dacc, false, false);

            // ---- bias + control conv + ReLU ----
            const int roff = lanelo ? 0 : 8;   // D rows: lanes<16 -> 0..7, else 8..15
            float hv[8];
            #pragma unroll
            for (int r = 0; r < 8; ++r) {
                const int tr = t0 + roff + r;
                float v = dacc[r] + bias + ctrls[tr] * cw + cbb;
                hv[r] = v > 0.0f ? v : 0.0f;
            }

            // ---- mixer accumulation over last 128 timesteps ----
            // (tiles 264..271, exactly one per wave 8..15 -> wave-uniform branch)
            if (t0 >= N_WIN - 128) {
                #pragma unroll
                for (int r = 0; r < 8; ++r) {
                    float v = hv[r] * mw;          // mw==0 on pad channels
                    v += __shfl_xor(v, 1);
                    v += __shfl_xor(v, 2);
                    v += __shfl_xor(v, 4);
                    v += __shfl_xor(v, 8);         // sum over 16-lane half
                    if (co == 0) {
                        oacc[t0 + roff + r - (N_WIN - 128)] += v;  // single owner
                    }
                }
            }

            if (!last) {
                // ---- stage h (D-layout -> row-major) for residual GEMM A ----
                // hstage[wave] is wave-private; per-wave LDS ops are in-order,
                // so a compiler scheduling fence is sufficient.
                if (realc) {
                    _Float16* hs = &hstage[wave][0];
                    #pragma unroll
                    for (int r = 0; r < 8; ++r)
                        hs[(roff + r) * 8 + co] = (_Float16)hv[r];
                }
                __builtin_amdgcn_wave_barrier();

                Frag16 a2;                       // A: K = channel o (0..7)
                if (lanelo) {
                    a2.f4[0] = *reinterpret_cast<const float4*>(&hstage[wave][m * 8]);
                    a2.f4[1] = make_float4(0.f, 0.f, 0.f, 0.f);
                } else {
                    a2.f4[0] = make_float4(0.f, 0.f, 0.f, 0.f);
                    a2.f4[1] = make_float4(0.f, 0.f, 0.f, 0.f);
                }
                v8f cacc;                        // C = x_cur + io_b
                const int cc = co & 7;
                #pragma unroll
                for (int r = 0; r < 8; ++r)
                    cacc[r] = (float)cur[(t0 + roff + r) * 8 + cc] + iob;

                v8f d2 = __builtin_amdgcn_wmma_f32_16x16x32_f16(
                             false, a2.v, false, bio.v, (short)0, cacc, false, false);

                if (realc) {
                    #pragma unroll
                    for (int r = 0; r < 8; ++r)
                        nxt[(t0 + roff + r) * 8 + co] = (_Float16)d2[r];
                }
                __builtin_amdgcn_wave_barrier();  // guard hstage reuse next tile
            }
        }
        __syncthreads();           // x_next complete before next layer reads it
        _Float16* tmp = cur; cur = nxt; nxt = tmp;
    }

    if (tid < 128) {
        out[b * 128 + tid] = oacc[tid] + mix_b[0];
    }
}

extern "C" void kernel_launch(void* const* d_in, const int* in_sizes, int n_in,
                              void* d_out, int out_size, void* d_ws, size_t ws_size,
                              hipStream_t stream) {
    (void)n_in; (void)out_size; (void)d_ws; (void)ws_size;
    const int B = in_sizes[0] / T_FULL;   // 32
    wavenet_kernel<<<B, NTHREADS, 0, stream>>>(
        (const float*)d_in[0],  // audio
        (const float*)d_in[1],  // ctrl
        (const float*)d_in[2],  // c_w0
        (const float*)d_in[3],  // c_ws
        (const float*)d_in[4],  // c_b
        (const float*)d_in[5],  // ctrl_w
        (const float*)d_in[6],  // ctrl_b
        (const float*)d_in[7],  // io_w
        (const float*)d_in[8],  // io_b
        (const float*)d_in[9],  // mix_w
        (const float*)d_in[10], // mix_b
        (float*)d_out);
}
